// DADA3d_58128087384154
// MI455X (gfx1250) — compile-verified, matrix operand order checked by reading, not run on hardware
//
#include <hip/hip_runtime.h>
#include <math.h>

typedef __attribute__((ext_vector_type(16))) _Float16 v16h;
typedef __attribute__((ext_vector_type(8)))  _Float16 h8;
typedef __attribute__((ext_vector_type(8)))  float    v8f;
typedef __attribute__((ext_vector_type(4)))  float    f4;
typedef __attribute__((ext_vector_type(4)))  int      i4;

#define C_DIM 128
#define K_NBR 64
#define H_HEADS 8
#define HD_DIM 16
#define EPS_BN 1e-5f

#if defined(__has_builtin)
#if __has_builtin(__builtin_amdgcn_global_load_async_to_lds_b128) && \
    __has_builtin(__builtin_amdgcn_s_wait_asynccnt)
#define HAVE_ASYNC_LDS 1
#endif
#endif

#ifdef HAVE_ASYNC_LDS
typedef __attribute__((address_space(1))) i4 gi4;   // global int4
typedef __attribute__((address_space(3))) i4 li4;   // LDS int4
#define ASYNC_CP_B128(gptr, lptr)                                              \
    __builtin_amdgcn_global_load_async_to_lds_b128((gi4*)(gptr), (li4*)(lptr), \
                                                   0, 0)
#endif

// ---------------------------------------------------------------------------
// gq16/gk16 = f16( vf + relu(coords @ {q,k}_pos_w^T + b) )
// ---------------------------------------------------------------------------
__global__ __launch_bounds__(256)
void posenc_kernel(const float* __restrict__ vf, const float* __restrict__ coords,
                   const float* __restrict__ qw, const float* __restrict__ qb,
                   const float* __restrict__ kw, const float* __restrict__ kb,
                   _Float16* __restrict__ gq16, _Float16* __restrict__ gk16, int N)
{
    int i = blockIdx.x * blockDim.x + threadIdx.x;
    if (i >= N * C_DIM) return;
    int n = i >> 7;
    int c = i & (C_DIM - 1);
    float x = coords[n * 3 + 0], y = coords[n * 3 + 1], z = coords[n * 3 + 2];
    float f = vf[i];
    float pq = qw[c * 3 + 0] * x + qw[c * 3 + 1] * y + qw[c * 3 + 2] * z + qb[c];
    float pk = kw[c * 3 + 0] * x + kw[c * 3 + 1] * y + kw[c * 3 + 2] * z + kb[c];
    gq16[i] = (_Float16)(f + fmaxf(pq, 0.f));
    gk16[i] = (_Float16)(f + fmaxf(pk, 0.f));
}

__global__ __launch_bounds__(256)
void f32_to_f16_kernel(const float* __restrict__ x, _Float16* __restrict__ y, int n)
{
    int i = blockIdx.x * blockDim.x + threadIdx.x;
    if (i < n) y[i] = (_Float16)x[i];
}

// ---------------------------------------------------------------------------
// WMMA GEMM, f16 operands, f32 accumulate.
// Y = act((X[M,Kd] @ W[Nout,Kd]^T + bias) * alpha + residual)
// Block = 128 threads (4 waves), tile 64(M) x 64(N), K-step 64 -> 8 wmma/iter.
// Async global->LDS staging; FULLM template removes tail guards from the
// interior-block fast path.
// ---------------------------------------------------------------------------
union AFrag { v16h v; f4 f[2]; };

template <bool FULLM>
__device__ __forceinline__ void gemm_tiles(
    const _Float16* __restrict__ X, const _Float16* __restrict__ W,
    _Float16 (*As)[72], _Float16 (*Bs)[72],
    int M, int Kd, int m_block, int n_block,
    int tid, int wave, int lane, v8f acc[4])
{
    const int hi   = lane >> 4;
    const int mrow = lane & 15;

    for (int k0 = 0; k0 < Kd; k0 += 64) {
        // stage 64x64 f16 tiles: 512 16B chunks, 4 per thread
#pragma unroll
        for (int j = 0; j < 4; ++j) {
            int id = tid + 128 * j;
            int r  = id >> 3;              // 0..63
            int c8 = (id & 7) * 8;         // 0,8,...,56 halves
            const _Float16* gA = X + (size_t)(m_block + r) * Kd + k0 + c8;
            const _Float16* gB = W + (size_t)(n_block + r) * Kd + k0 + c8;
#ifdef HAVE_ASYNC_LDS
            ASYNC_CP_B128(gB, &Bs[r][c8]);
            if (FULLM) {
                ASYNC_CP_B128(gA, &As[r][c8]);
            } else {
                h8 av = {};
                if (m_block + r < M) av = *(const h8*)gA;
                *(h8*)&As[r][c8] = av;
            }
#else
            *(h8*)&Bs[r][c8] = *(const h8*)gB;
            h8 av = {};
            if (FULLM || (m_block + r < M)) av = *(const h8*)gA;
            *(h8*)&As[r][c8] = av;
#endif
        }
        if (k0 + 64 < Kd) {  // prefetch next K tile
            __builtin_prefetch(X + (size_t)(m_block + (tid >> 3)) * Kd + k0 + 64, 0, 1);
            __builtin_prefetch(W + (size_t)(n_block + (tid >> 3)) * Kd + k0 + 64, 0, 1);
        }
#ifdef HAVE_ASYNC_LDS
        __builtin_amdgcn_s_wait_asynccnt(0);
#endif
        __syncthreads();

        // batch all fragment loads, then a clean 8-deep WMMA burst
        const int am = 16 * wave + mrow;
        AFrag a[2];
        AFrag b[2][4];
#pragma unroll
        for (int s = 0; s < 2; ++s) {
            const int ks = s * 32;
            a[s].f[0] = *(const f4*)&As[am][ks + hi * 8];
            a[s].f[1] = *(const f4*)&As[am][ks + 16 + hi * 8];
#pragma unroll
            for (int t = 0; t < 4; ++t) {
                const int bn = 16 * t + mrow;
                b[s][t].f[0] = *(const f4*)&Bs[bn][ks + hi * 8];
                b[s][t].f[1] = *(const f4*)&Bs[bn][ks + 16 + hi * 8];
            }
        }
#pragma unroll
        for (int s = 0; s < 2; ++s) {
#pragma unroll
            for (int t = 0; t < 4; ++t) {
                acc[t] = __builtin_amdgcn_wmma_f32_16x16x32_f16(
                    false, a[s].v, false, b[s][t].v, (short)0, acc[t], false, false);
            }
        }
        __syncthreads();
    }
}

__global__ __launch_bounds__(128)
void wmma_gemm_f16(const _Float16* __restrict__ X, const _Float16* __restrict__ W,
                   const float* __restrict__ bias, const float* __restrict__ residual,
                   float* __restrict__ Yf, _Float16* __restrict__ Yh,
                   int M, int Kd, int Nout, float alpha, int do_relu)
{
    __shared__ _Float16 As[64][72];   // 64 x 64K halves (+8 pad -> 144B stride)
    __shared__ _Float16 Bs[64][72];

    const int tid  = threadIdx.x;
    const int wave = tid >> 5;
    const int lane = tid & 31;
    const int hi   = lane >> 4;
    const int mrow = lane & 15;
    const int m_block = blockIdx.x * 64;
    const int n_block = blockIdx.y * 64;

    v8f acc[4];
    v8f zero8 = {0.f, 0.f, 0.f, 0.f, 0.f, 0.f, 0.f, 0.f};
#pragma unroll
    for (int t = 0; t < 4; ++t) acc[t] = zero8;

    if (m_block + 64 <= M) {
        gemm_tiles<true >(X, W, As, Bs, M, Kd, m_block, n_block, tid, wave, lane, acc);
    } else {
        gemm_tiles<false>(X, W, As, Bs, M, Kd, m_block, n_block, tid, wave, lane, acc);
    }

    // epilogue: vgpr r holds M = 16*wave + hi*8 + r, N = n_block + 16*t + mrow
#pragma unroll
    for (int t = 0; t < 4; ++t) {
        int gn = n_block + 16 * t + mrow;
        float bv = bias[gn];
#pragma unroll
        for (int r = 0; r < 8; ++r) {
            int gm = m_block + 16 * wave + hi * 8 + r;
            if (gm < M) {
                float val = (acc[t][r] + bv) * alpha;
                if (residual) val += residual[(size_t)gm * Nout + gn];
                if (do_relu)  val  = fmaxf(val, 0.f);
                if (Yf) Yf[(size_t)gm * Nout + gn] = val;
                if (Yh) Yh[(size_t)gm * Nout + gn] = (_Float16)val;
            }
        }
    }
}

// ---------------------------------------------------------------------------
// Attention: one block (128 threads) per voxel; Q pre-scaled; f16 output.
// ---------------------------------------------------------------------------
__global__ __launch_bounds__(128)
void attn_kernel(const float* __restrict__ Q, const float* __restrict__ Km,
                 const float* __restrict__ Vm, const int* __restrict__ idx,
                 const unsigned char* __restrict__ mask, _Float16* __restrict__ att16)
{
    __shared__ float qs[C_DIM];
    __shared__ int   sidx[K_NBR];
    __shared__ float sc[H_HEADS][K_NBR];
    __shared__ float aw[H_HEADS][K_NBR];

    const int n   = blockIdx.x;
    const int tid = threadIdx.x;

    qs[tid] = Q[(size_t)n * C_DIM + tid];
    if (tid < K_NBR) sidx[tid] = idx[(size_t)n * K_NBR + tid];
    __syncthreads();

#pragma unroll
    for (int i = 0; i < 4; ++i) {
        int p = tid + 128 * i;
        int k = p & (K_NBR - 1);
        int h = p >> 6;
        int j = sidx[k];
        const f4* kp = (const f4*)(Km + (size_t)j * C_DIM + h * HD_DIM);
        const f4* qp = (const f4*)(qs + h * HD_DIM);
        float s = 0.f;
#pragma unroll
        for (int q4 = 0; q4 < 4; ++q4) {
            f4 kv = kp[q4], qv = qp[q4];
            s += kv.x * qv.x + kv.y * qv.y + kv.z * qv.z + kv.w * qv.w;
        }
        sc[h][k] = mask[(size_t)n * K_NBR + k] ? -1e30f : s;
    }
    __syncthreads();

    if (tid < H_HEADS) {
        float mx = -1e30f;
        for (int k = 0; k < K_NBR; ++k) mx = fmaxf(mx, sc[tid][k]);
        float sum = 0.f;
        for (int k = 0; k < K_NBR; ++k) {
            float e = __expf(sc[tid][k] - mx);
            aw[tid][k] = e;
            sum += e;
        }
        float inv = 1.f / sum;
        for (int k = 0; k < K_NBR; ++k) aw[tid][k] *= inv;
    }
    __syncthreads();

    const int h = tid >> 4;
    float acc = 0.f;
    for (int k = 0; k < K_NBR; ++k) {
        acc += aw[h][k] * Vm[(size_t)sidx[k] * C_DIM + tid];
    }
    att16[(size_t)n * C_DIM + tid] = (_Float16)acc;
}

// ---------------------------------------------------------------------------
// BatchNorm (training-mode batch stats over N)
// ---------------------------------------------------------------------------
__global__ __launch_bounds__(256)
void bn_reduce_kernel(const float* __restrict__ x, float* __restrict__ stats,
                      int Nr, int Cc)
{
    __shared__ float ss[256], s2s[256];
    const int c   = blockIdx.x;
    const int tid = threadIdx.x;
    float s = 0.f, s2 = 0.f;
    for (int r = tid; r < Nr; r += 256) {
        float v = x[(size_t)r * Cc + c];
        s += v; s2 += v * v;
    }
    ss[tid] = s; s2s[tid] = s2;
    __syncthreads();
    for (int off = 128; off > 0; off >>= 1) {
        if (tid < off) { ss[tid] += ss[tid + off]; s2s[tid] += s2s[tid + off]; }
        __syncthreads();
    }
    if (tid == 0) { stats[2 * c] = ss[0]; stats[2 * c + 1] = s2s[0]; }
}

__global__ __launch_bounds__(256)
void bn_apply_kernel(const float* __restrict__ x, const float* __restrict__ stats,
                     const float* __restrict__ g, const float* __restrict__ b,
                     float* __restrict__ yf, _Float16* __restrict__ yh,
                     int Nr, int Cc, int do_relu)
{
    int i = blockIdx.x * blockDim.x + threadIdx.x;
    if (i >= Nr * Cc) return;
    int c = i % Cc;
    float invN  = 1.f / (float)Nr;
    float mean  = stats[2 * c] * invN;
    float var   = stats[2 * c + 1] * invN - mean * mean;
    float v = (x[i] - mean) * rsqrtf(var + EPS_BN) * g[c] + b[c];
    if (do_relu) v = fmaxf(v, 0.f);
    if (yf) yf[i] = v;
    if (yh) yh[i] = (_Float16)v;
}

// ---------------------------------------------------------------------------
extern "C" void kernel_launch(void* const* d_in, const int* in_sizes, int n_in,
                              void* d_out, int out_size, void* d_ws, size_t ws_size,
                              hipStream_t stream)
{
    const int C = 128, FF = 512, OUT = 128;
    const int N = in_sizes[0] / C;

    const float* vf         = (const float*)d_in[0];
    const float* coords     = (const float*)d_in[1];
    const int*   key_idx    = (const int*)d_in[2];
    const unsigned char* km = (const unsigned char*)d_in[3];
    const float* in_proj_w  = (const float*)d_in[4];
    const float* in_proj_b  = (const float*)d_in[5];
    const float* out_proj_w = (const float*)d_in[6];
    const float* out_proj_b = (const float*)d_in[7];
    const float* q_pos_w    = (const float*)d_in[8];
    const float* q_pos_b    = (const float*)d_in[9];
    const float* k_pos_w    = (const float*)d_in[10];
    const float* k_pos_b    = (const float*)d_in[11];
    const float* lin1_w     = (const float*)d_in[12];
    const float* lin1_b     = (const float*)d_in[13];
    const float* lin2_w     = (const float*)d_in[14];
    const float* lin2_b     = (const float*)d_in[15];
    const float* bn1_g      = (const float*)d_in[16];
    const float* bn1_b      = (const float*)d_in[17];
    const float* bn2_g      = (const float*)d_in[18];
    const float* bn2_b      = (const float*)d_in[19];
    const float* out_w      = (const float*)d_in[20];
    const float* out_b      = (const float*)d_in[21];
    const float* bn3_g      = (const float*)d_in[22];
    const float* bn3_b      = (const float*)d_in[23];

    const size_t NC = (size_t)N * C;
    const size_t NF = (size_t)N * FF;

    // f32 region
    float* ws    = (float*)d_ws;
    float* Qb    = ws;
    float* Kmat  = Qb + NC;
    float* Vmat  = Kmat + NC;
    float* x1raw = Vmat + NC;
    float* x1    = x1raw + NC;
    float* x2raw = x1 + NC;
    float* y3    = x2raw + NC;
    float* st1   = y3 + NC;
    float* st2   = st1 + 2 * C;
    float* st3   = st2 + 2 * C;

    // f16 region
    _Float16* hb     = (_Float16*)(st3 + 2 * OUT);
    _Float16* gq16   = hb;
    _Float16* gk16   = gq16 + NC;
    _Float16* att16  = gk16 + NC;
    _Float16* x1_16  = att16 + NC;
    _Float16* x2_16  = x1_16 + NC;
    _Float16* ffh16  = x2_16 + NC;            // [N, FF]
    _Float16* wqkv16 = ffh16 + NF;            // 3*C*C
    _Float16* outp16 = wqkv16 + 3 * C * C;    // C*C
    _Float16* l1w16  = outp16 + C * C;        // FF*C
    _Float16* l2w16  = l1w16 + FF * C;        // C*FF
    _Float16* ow16   = l2w16 + C * FF;        // OUT*C

    const float scale = 0.25f;                // 1/sqrt(HD=16)
    dim3 b256(256), b128(128);
    dim3 gElem((N * C + 255) / 256);
    dim3 gG1((N + 63) / 64, C / 64);
    dim3 gG2((N + 63) / 64, FF / 64);
    dim3 gG3((N + 63) / 64, OUT / 64);

    // weight down-conversion (f32 -> f16), once per launch
    f32_to_f16_kernel<<<dim3((3 * C * C + 255) / 256), b256, 0, stream>>>(in_proj_w, wqkv16, 3 * C * C);
    f32_to_f16_kernel<<<dim3((C * C + 255) / 256),     b256, 0, stream>>>(out_proj_w, outp16, C * C);
    f32_to_f16_kernel<<<dim3((FF * C + 255) / 256),    b256, 0, stream>>>(lin1_w, l1w16, FF * C);
    f32_to_f16_kernel<<<dim3((C * FF + 255) / 256),    b256, 0, stream>>>(lin2_w, l2w16, C * FF);
    f32_to_f16_kernel<<<dim3((OUT * C + 255) / 256),   b256, 0, stream>>>(out_w, ow16, OUT * C);

    // 1. positional embedding fusion (f16 outputs)
    posenc_kernel<<<gElem, b256, 0, stream>>>(vf, coords, q_pos_w, q_pos_b,
                                              k_pos_w, k_pos_b, gq16, gk16, N);

    // 2. Q/K/V projections (WMMA)
    wmma_gemm_f16<<<gG1, b128, 0, stream>>>(gq16, wqkv16,             in_proj_b,
                                            nullptr, Qb, nullptr,   N, C, C, scale, 0);
    wmma_gemm_f16<<<gG1, b128, 0, stream>>>(gk16, wqkv16 + 1 * C * C, in_proj_b + C,
                                            nullptr, Kmat, nullptr, N, C, C, 1.f,   0);
    wmma_gemm_f16<<<gG1, b128, 0, stream>>>(gk16, wqkv16 + 2 * C * C, in_proj_b + 2 * C,
                                            nullptr, Vmat, nullptr, N, C, C, 1.f,   0);

    // 3. masked neighborhood attention
    attn_kernel<<<dim3(N), b128, 0, stream>>>(Qb, Kmat, Vmat, key_idx, km, att16);

    // 4. out_proj + residual(vf) -> BN1
    wmma_gemm_f16<<<gG1, b128, 0, stream>>>(att16, outp16, out_proj_b,
                                            vf, x1raw, nullptr, N, C, C, 1.f, 0);
    bn_reduce_kernel<<<dim3(C), b256, 0, stream>>>(x1raw, st1, N, C);
    bn_apply_kernel<<<gElem, b256, 0, stream>>>(x1raw, st1, bn1_g, bn1_b,
                                                x1, x1_16, N, C, 0);

    // 5. FFN + residual(x1) -> BN2
    wmma_gemm_f16<<<gG2, b128, 0, stream>>>(x1_16, l1w16, lin1_b,
                                            nullptr, nullptr, ffh16, N, C, FF, 1.f, 1);
    wmma_gemm_f16<<<gG1, b128, 0, stream>>>(ffh16, l2w16, lin2_b,
                                            x1, x2raw, nullptr, N, FF, C, 1.f, 0);
    bn_reduce_kernel<<<dim3(C), b256, 0, stream>>>(x2raw, st2, N, C);
    bn_apply_kernel<<<gElem, b256, 0, stream>>>(x2raw, st2, bn2_g, bn2_b,
                                                nullptr, x2_16, N, C, 0);

    // 6. output head -> BN3 -> relu -> d_out
    wmma_gemm_f16<<<gG3, b128, 0, stream>>>(x2_16, ow16, out_b,
                                            nullptr, y3, nullptr, N, C, OUT, 1.f, 0);
    bn_reduce_kernel<<<dim3(OUT), b256, 0, stream>>>(y3, st3, N, OUT);
    dim3 gOut((N * OUT + 255) / 256);
    bn_apply_kernel<<<gOut, b256, 0, stream>>>(y3, st3, bn3_g, bn3_b,
                                               (float*)d_out, nullptr, N, OUT, 1);
}
